// NTXentGraphLoss_90606630076757
// MI455X (gfx1250) — compile-verified, matrix operand order checked by reading, not run on hardware
//
#include <hip/hip_runtime.h>
#include <hip/hip_bf16.h>

typedef __attribute__((ext_vector_type(16))) _Float16 v16h;
typedef __attribute__((ext_vector_type(8)))  float    v8f;

#define BROWS  4096
#define DIM    512
#define NROW   8192              // 2B rows
#define NT     (NROW / 16)       // 512 column tiles
#define KK     (DIM / 32)        // 16 k-steps of 32
#define NSPLIT 8                 // N-dimension split across blocks
#define NTSEG  (NT / NSPLIT)     // 64 column tiles per split
#define LDSP   520               // padded LDS row stride in halves (4-bank skew)
#define INVT   10.0f             // 1 / temperature
#define PIPE   3                 // B-fragment register pipeline depth

// ---------------------------------------------------------------------------
// Kernel 1: L2-normalize rows of [z_i ; z_j] and emit f16 matrix zh (8192x512)
// ---------------------------------------------------------------------------
__global__ __launch_bounds__(256)
void ntx_norm(const float* __restrict__ zi, const float* __restrict__ zj,
              _Float16* __restrict__ zh) {
  const int wave = threadIdx.x >> 5;
  const int lane = threadIdx.x & 31;
  const int row  = blockIdx.x * 8 + wave;          // 1024 blocks * 8 waves
  const float* src = (row < BROWS) ? (zi + (size_t)row * DIM)
                                   : (zj + (size_t)(row - BROWS) * DIM);
  float4 d[4];
  float s = 0.0f;
#pragma unroll
  for (int k = 0; k < 4; ++k) {
    d[k] = ((const float4*)src)[lane + k * 32];
    s += d[k].x * d[k].x + d[k].y * d[k].y + d[k].z * d[k].z + d[k].w * d[k].w;
  }
#pragma unroll
  for (int off = 16; off > 0; off >>= 1) s += __shfl_xor(s, off);
  const float inv = 1.0f / fmaxf(sqrtf(s), 1e-12f);

  _Float16* dst = zh + (size_t)row * DIM;
  union H4 { _Float16 h[4]; uint2 u; };
#pragma unroll
  for (int k = 0; k < 4; ++k) {
    H4 o;
    o.h[0] = (_Float16)(d[k].x * inv);
    o.h[1] = (_Float16)(d[k].y * inv);
    o.h[2] = (_Float16)(d[k].z * inv);
    o.h[3] = (_Float16)(d[k].w * inv);
    ((uint2*)dst)[lane + k * 32] = o.u;
  }
}

// ---------------------------------------------------------------------------
// Kernel 2: fused WMMA GEMM + streaming exp-sum (fixed max = 1/T) + pos-logit
// grid = NSPLIT * 64 blocks, 256 threads (8 waves, one 16-row tile per wave)
// ---------------------------------------------------------------------------
__global__ __launch_bounds__(256) __attribute__((amdgpu_waves_per_eu(1)))
void ntx_lse(const _Float16* __restrict__ zh,
             float* __restrict__ sumpart,          // [NSPLIT][NROW]
             float* __restrict__ poslog) {         // [NROW]
  __shared__ __align__(16) _Float16 panel[2][16 * LDSP];

  const int tid    = threadIdx.x;
  const int wave   = tid >> 5;
  const int lane   = tid & 31;
  const int hl     = lane >> 4;                    // half-wave select
  const int ln     = lane & 15;                    // row/col within tile
  const int rowblk = blockIdx.x & 63;
  const int split  = blockIdx.x >> 6;              // 0..NSPLIT-1
  const int r0     = rowblk * 128 + wave * 16;     // this wave's 16 rows
  const int ntlo   = split * NTSEG;
  const int nthi   = ntlo + NTSEG;

  // ---- preload all A fragments for 16 rows x K=512 (16 x v16h = 128 VGPRs)
  const _Float16* abase = zh + (size_t)(r0 + ln) * DIM + (hl << 3);
  v16h a[KK];
#pragma unroll
  for (int kk = 0; kk < KK; ++kk) {
    union { uint4 u[2]; v16h h; } x;
    x.u[0] = *(const uint4*)(abase + kk * 32);
    x.u[1] = *(const uint4*)(abase + kk * 32 + 16);
    a[kk] = x.h;
  }

  // ---- cooperative panel staging map: 256 threads, 16 rows x 1KB each
  const int srow = tid >> 4;                       // 0..15 : column-row of panel
  const int scol = tid & 15;                       // 0..15 : 64B chunk
  {
    const _Float16* g = zh + (size_t)(ntlo * 16 + srow) * DIM + scol * 32;
    uint4 p0 = ((const uint4*)g)[0];
    uint4 p1 = ((const uint4*)g)[1];
    uint4 p2 = ((const uint4*)g)[2];
    uint4 p3 = ((const uint4*)g)[3];
    uint4* l = (uint4*)(&panel[0][srow * LDSP + scol * 32]);
    l[0] = p0; l[1] = p1; l[2] = p2; l[3] = p3;
  }
  __syncthreads();

  const int ntp = ((r0 + BROWS) & (NROW - 1)) >> 4;  // tile holding pos pair
  float acc[8] = {0.f, 0.f, 0.f, 0.f, 0.f, 0.f, 0.f, 0.f};

  for (int nt = ntlo; nt < nthi; ++nt) {
    // prefetch next panel into registers (overlaps with WMMA below)
    const int np = (nt + 1 < nthi) ? nt + 1 : nthi - 1;
    const _Float16* g = zh + (size_t)(np * 16 + srow) * DIM + scol * 32;
    uint4 p0 = ((const uint4*)g)[0];
    uint4 p1 = ((const uint4*)g)[1];
    uint4 p2 = ((const uint4*)g)[2];
    uint4 p3 = ((const uint4*)g)[3];

    // ---- 16x16 logits tile: 16 WMMA, B fragments pipelined PIPE deep
    const _Float16* bb = &panel[nt & 1][(size_t)ln * LDSP + (hl << 3)];
    union F { uint4 u[2]; v16h h; };
    F bf[PIPE];
#pragma unroll
    for (int i = 0; i < PIPE; ++i) {
      bf[i].u[0] = *(const uint4*)(bb + i * 32);
      bf[i].u[1] = *(const uint4*)(bb + i * 32 + 16);
    }
    v8f cA = {0.f, 0.f, 0.f, 0.f, 0.f, 0.f, 0.f, 0.f};
    v8f cB = {0.f, 0.f, 0.f, 0.f, 0.f, 0.f, 0.f, 0.f};
#pragma unroll
    for (int kk = 0; kk < KK; ++kk) {
      v16h bh = bf[kk % PIPE].h;
      if (kk + PIPE < KK) {                        // refill ring slot
        bf[kk % PIPE].u[0] = *(const uint4*)(bb + (kk + PIPE) * 32);
        bf[kk % PIPE].u[1] = *(const uint4*)(bb + (kk + PIPE) * 32 + 16);
      }
      if (kk & 1)
        cB = __builtin_amdgcn_wmma_f32_16x16x32_f16(false, a[kk], false, bh,
                                                    (short)0, cB, false, false);
      else
        cA = __builtin_amdgcn_wmma_f32_16x16x32_f16(false, a[kk], false, bh,
                                                    (short)0, cA, false, false);
    }
    // ---- pin the software pipeline: 6 DS-reads, then (1 WMMA, 2 DS-reads)x13,
    //      then 3 WMMAs.  0x008 = MFMA/WMMA group, 0x100 = DS read.
    __builtin_amdgcn_sched_group_barrier(0x100, 2 * PIPE, 0);
#pragma unroll
    for (int i = 0; i < KK - PIPE; ++i) {
      __builtin_amdgcn_sched_group_barrier(0x008, 1, 0);
      __builtin_amdgcn_sched_group_barrier(0x100, 2, 0);
    }
    __builtin_amdgcn_sched_group_barrier(0x008, PIPE, 0);

    const int  ct    = nt << 4;
    const bool diagT = (ct == r0);
    const bool posT  = (nt == ntp);
#pragma unroll
    for (int v = 0; v < 8; ++v) {
      const int  m   = v + (hl << 3);              // row within tile
      const float lg = (cA[v] + cB[v]) * INVT;     // logit
      float e = __expf(lg - INVT);                 // exp(l - max), max = 1/T
      const bool md = (m == ln);
      if (diagT && md) e = 0.0f;                   // mask self-similarity
      acc[v] += e;
      if (posT && md) poslog[r0 + m] = lg;         // positive-pair logit
    }

    // commit prefetched panel and sync (single barrier per iteration)
    uint4* l = (uint4*)(&panel[(nt + 1) & 1][srow * LDSP + scol * 32]);
    l[0] = p0; l[1] = p1; l[2] = p2; l[3] = p3;
    __syncthreads();
  }

  // reduce exp-sums across the 16 columns held by each half-wave
#pragma unroll
  for (int v = 0; v < 8; ++v) {
    float s = acc[v];
    s += __shfl_xor(s, 1);
    s += __shfl_xor(s, 2);
    s += __shfl_xor(s, 4);
    s += __shfl_xor(s, 8);
    if (ln == 0) sumpart[(size_t)split * NROW + r0 + v + (hl << 3)] = s;
  }
}

// ---------------------------------------------------------------------------
// Kernel 3: combine N-splits, finish logsumexp, reduce loss
// ---------------------------------------------------------------------------
__global__ __launch_bounds__(256)
void ntx_loss(const float* __restrict__ sumpart, const float* __restrict__ poslog,
              float* __restrict__ out) {
  __shared__ float red[256];
  float s = 0.0f;
  for (int i = threadIdx.x; i < NROW; i += 256) {
    float t = 0.0f;
#pragma unroll
    for (int p = 0; p < NSPLIT; ++p) t += sumpart[(size_t)p * NROW + i];
    s += (INVT + __logf(t)) - poslog[i];           // lse_i - pos_i
  }
  red[threadIdx.x] = s;
  __syncthreads();
  for (int o = 128; o > 0; o >>= 1) {
    if (threadIdx.x < o) red[threadIdx.x] += red[threadIdx.x + o];
    __syncthreads();
  }
  if (threadIdx.x == 0) out[0] = red[0] / (float)NROW;
}

// ---------------------------------------------------------------------------
extern "C" void kernel_launch(void* const* d_in, const int* in_sizes, int n_in,
                              void* d_out, int out_size, void* d_ws, size_t ws_size,
                              hipStream_t stream) {
  const float* zi = (const float*)d_in[0];
  const float* zj = (const float*)d_in[1];

  char* ws = (char*)d_ws;
  _Float16* zh      = (_Float16*)ws;                                  // 8 MB
  float*    poslog  = (float*)(ws + (size_t)NROW * DIM * 2);          // 32 KB
  float*    sumpart = (float*)(ws + (size_t)NROW * DIM * 2 + NROW*4); // 256 KB

  ntx_norm<<<NROW / 8, 256, 0, stream>>>(zi, zj, zh);
  ntx_lse <<<NSPLIT * 64, 256, 0, stream>>>(zh, sumpart, poslog);
  ntx_loss<<<1, 256, 0, stream>>>(sumpart, poslog, (float*)d_out);
}
